// HANLayer_14491219656747
// MI455X (gfx1250) — matrix operand order repaired and checked by reading, not compiled, not run on hardware
//
#include <hip/hip_runtime.h>
#include <hip/hip_bf16.h>
#include <math.h>

// ---------------------------------------------------------------------------
// HANLayer for MI455X (gfx1250, wave32).
//  - feat = h @ W[m]   : split-bf16 WMMA; A/B pre-swizzled into fragment
//                        layout (hi/lo) so the inner loop is loads + wmma only.
//  - edge softmax/aggr : uint-encoded atomicMax + f32 atomics (rst fits L2).
//  - semantic MLP      : split-bf16 WMMA (A built on the fly with fused ELU,
//                        B pre-swizzled), tanh + shfl reduction + atomicAdd.
// Workspace ~290 MB.
// ---------------------------------------------------------------------------

#define NN   50000
#define EE   800000
#define MMP  2
#define INF_ 256
#define HH   4
#define DD   64
#define HDIM 256
#define SHID 128
#define SLOPE 0.2f

typedef __attribute__((ext_vector_type(16))) __bf16          v16bf;
typedef __attribute__((ext_vector_type(16))) unsigned short  v16us;
typedef __attribute__((ext_vector_type(8)))  float           v8f;

__device__ inline float eluf(float x)   { return x > 0.0f ? x : expm1f(x); }
__device__ inline float lreluf(float x) { return x > 0.0f ? x : SLOPE * x; }

// monotonic uint encoding of float (for atomicMax over f32)
__device__ inline unsigned fenc(float f) {
  unsigned u = __float_as_uint(f);
  return (u & 0x80000000u) ? ~u : (u | 0x80000000u);
}
__device__ inline float fdec(unsigned v) {
  unsigned b = (v & 0x80000000u) ? (v & 0x7fffffffu) : ~v;
  return __uint_as_float(b);
}

// split a float into bf16 hi + bf16 lo (native v_cvt, RNE)
__device__ inline void bf16_split(float x, unsigned short& h, unsigned short& l) {
  __bf16 hb = (__bf16)x;
  h = __builtin_bit_cast(unsigned short, hb);
  l = __builtin_bit_cast(unsigned short, (__bf16)(x - (float)hb));
}

__device__ inline v16bf ldfrag(const unsigned short* __restrict__ p) {
  return __builtin_bit_cast(v16bf, *(const v16us*)p);
}

// split-bf16 product: a*b ~= ah*bh + ah*bl + al*bh  (~2^-17 rel, f32 accum)
__device__ inline v8f wmma3(v8f acc, v16bf ah, v16bf al, v16bf bh, v16bf bl) {
  acc = __builtin_amdgcn_wmma_f32_16x16x32_bf16(false, ah, false, bh, (short)0, acc, false, false);
  acc = __builtin_amdgcn_wmma_f32_16x16x32_bf16(false, ah, false, bl, (short)0, acc, false, false);
  acc = __builtin_amdgcn_wmma_f32_16x16x32_bf16(false, al, false, bh, (short)0, acc, false, false);
  return acc;
}

// ---- pre-swizzle: row-major [R,K] f32 -> A-fragment records (hi/lo bf16) --
// record index ((rt*KT + kt)*32 + lane), 16 bf16 per record (32 B).
// A 16x32 layout (ISA 7.12.2): elem i: j=i>>1,b=i&1;
//   k = (j<4 ? 2j : 16+2(j-4)) + 8*(lane>>4) + b ; row = lane&15.
__global__ void han_swizzle_a(const float* __restrict__ src,
                              unsigned short* __restrict__ hi,
                              unsigned short* __restrict__ lo,
                              int R, int K) {
  const int KT = K >> 5;
  const int total = (R >> 4) * KT * 32;
  const int stride = gridDim.x * blockDim.x;
  for (int t = blockIdx.x * blockDim.x + threadIdx.x; t < total; t += stride) {
    const int lane = t & 31;
    const int kt = (t >> 5) % KT;
    const int rt = (t >> 5) / KT;
    const int half = lane >> 4;
    const int r = rt * 16 + (lane & 15);
    v16us hv, lv;
#pragma unroll
    for (int i = 0; i < 16; ++i) {
      const int j = i >> 1, b = i & 1;
      const int k = kt * 32 + ((j < 4) ? 2 * j : 16 + 2 * (j - 4)) + 8 * half + b;
      unsigned short hb, lb;
      bf16_split(src[(size_t)r * K + k], hb, lb);
      hv[i] = hb; lv[i] = lb;
    }
    *(v16us*)(hi + (size_t)t * 16) = hv;
    *(v16us*)(lo + (size_t)t * 16) = lv;
  }
}

// ---- pre-swizzle: row-major [K,Nc] f32 -> B-fragment records (hi/lo) ------
// record index ((kt*CT + ct)*32 + lane); elem i: k = kt*32 + (i<8 ? i : 16+(i-8)) + 8*(lane>>4),
// col = ct*16 + (lane&15).
__global__ void han_swizzle_b(const float* __restrict__ src,
                              unsigned short* __restrict__ hi,
                              unsigned short* __restrict__ lo,
                              int K, int Nc) {
  const int CT = Nc >> 4;
  const int total = (K >> 5) * CT * 32;
  const int stride = gridDim.x * blockDim.x;
  for (int t = blockIdx.x * blockDim.x + threadIdx.x; t < total; t += stride) {
    const int lane = t & 31;
    const int ct = (t >> 5) % CT;
    const int kt = (t >> 5) / CT;
    const int half = lane >> 4;
    const int c = ct * 16 + (lane & 15);
    v16us hv, lv;
#pragma unroll
    for (int i = 0; i < 16; ++i) {
      const int k = kt * 32 + ((i < 8) ? i : 16 + (i - 8)) + 8 * half;
      unsigned short hb, lb;
      bf16_split(src[(size_t)k * Nc + c], hb, lb);
      hv[i] = hb; lv[i] = lb;
    }
    *(v16us*)(hi + (size_t)t * 16) = hv;
    *(v16us*)(lo + (size_t)t * 16) = lv;
  }
}

// ---- K1: feat[m] = h @ W[m] from pre-swizzled fragments -------------------
// wave-task = (m, rowTile, colGroup of 4 tiles); inner loop: 8 frag loads
// + 12 v_wmma per k-step (A fragment amortized over 4 column tiles).
__global__ void han_feat_gemm(const unsigned short* __restrict__ Ahi,
                              const unsigned short* __restrict__ Alo,
                              const unsigned short* __restrict__ Bhi,
                              const unsigned short* __restrict__ Blo,
                              float* __restrict__ feat) {
  const int lane  = threadIdx.x & 31;
  const int wpb   = blockDim.x >> 5;
  int wave        = blockIdx.x * wpb + (threadIdx.x >> 5);
  const int nWave = gridDim.x * wpb;
  const int rowT = NN / 16, KT = INF_ / 32, CT = HDIM / 16, CG = CT / 4;
  const int total = MMP * rowT * CG;
  for (int t = wave; t < total; t += nWave) {
    const int m   = t / (rowT * CG);
    const int rem = t - m * rowT * CG;
    const int rt  = rem / CG;
    const int cg  = rem - rt * CG;
    const size_t bbase = (size_t)m * KT * CT * 32 * 16;
    v8f acc[4] = {{}, {}, {}, {}};
    for (int kt = 0; kt < KT; ++kt) {
      const size_t arec = ((size_t)(rt * KT + kt) * 32 + lane) * 16;
      const v16bf ah = ldfrag(Ahi + arec);
      const v16bf al = ldfrag(Alo + arec);
#pragma unroll
      for (int q = 0; q < 4; ++q) {
        const int ct = cg * 4 + q;
        const size_t brec = bbase + ((size_t)(kt * CT + ct) * 32 + lane) * 16;
        acc[q] = wmma3(acc[q], ah, al, ldfrag(Bhi + brec), ldfrag(Blo + brec));
      }
    }
    const int half = (lane >> 4) & 1;
    float* outm = feat + (size_t)m * NN * HDIM;
#pragma unroll
    for (int q = 0; q < 4; ++q) {
      const int c = (cg * 4 + q) * 16 + (lane & 15);
#pragma unroll
      for (int i = 0; i < 8; ++i)
        outm[(size_t)(rt * 16 + i + 8 * half) * HDIM + c] = acc[q][i];
    }
  }
}

// ---- K2: el/er per node (wave per (m,n), coalesced row read) --------------
__global__ void han_attn_scores(const float* __restrict__ feat,
                                const float* __restrict__ attn_l,
                                const float* __restrict__ attn_r,
                                float* __restrict__ el, float* __restrict__ er) {
  const int lane  = threadIdx.x & 31;
  const int wpb   = blockDim.x >> 5;
  int wave        = blockIdx.x * wpb + (threadIdx.x >> 5);
  const int nWave = gridDim.x * wpb;
  for (int t = wave; t < MMP * NN; t += nWave) {
    const int m = t / NN, n = t - m * NN;
    const float* f  = feat + ((size_t)m * NN + n) * HDIM;
    const float* al = attn_l + (size_t)m * HDIM;  // [H,D] flat: idx = head*64+d
    const float* ar = attn_r + (size_t)m * HDIM;
    float pl[HH] = {0, 0, 0, 0}, pr[HH] = {0, 0, 0, 0};
#pragma unroll
    for (int j = 0; j < 8; ++j) {
      const int c = lane + 32 * j;          // head = j>>1 (lane-invariant)
      const float v = f[c];
      pl[j >> 1] += v * al[c];
      pr[j >> 1] += v * ar[c];
    }
#pragma unroll
    for (int hh = 0; hh < HH; ++hh) {
      float a = pl[hh], b = pr[hh];
#pragma unroll
      for (int s = 16; s > 0; s >>= 1) {
        a += __shfl_xor(a, s, 32);
        b += __shfl_xor(b, s, 32);
      }
      if (lane == 0) {
        el[((size_t)m * NN + n) * HH + hh] = a;
        er[((size_t)m * NN + n) * HH + hh] = b;
      }
    }
  }
}

// ---- K3: per-dst max of leaky_relu(el[s]+er[d]) ---------------------------
__global__ void han_edge_max(const int* __restrict__ src, const int* __restrict__ dst,
                             const float* __restrict__ el, const float* __restrict__ er,
                             unsigned* __restrict__ nmax) {
  const int stride = gridDim.x * blockDim.x;
  for (int t = blockIdx.x * blockDim.x + threadIdx.x; t < MMP * EE; t += stride) {
    const int m = t / EE, e = t - m * EE;
    const int s = src[(size_t)m * EE + e];
    const int d = dst[(size_t)m * EE + e];
#pragma unroll
    for (int hh = 0; hh < HH; ++hh) {
      const float v = lreluf(el[((size_t)m * NN + s) * HH + hh] +
                             er[((size_t)m * NN + d) * HH + hh]);
      atomicMax(&nmax[((size_t)m * NN + d) * HH + hh], fenc(v));
    }
  }
}

// ---- K4: ex = exp(e - max[d]);  den[d] += ex ------------------------------
__global__ void han_edge_exp(const int* __restrict__ src, const int* __restrict__ dst,
                             const float* __restrict__ el, const float* __restrict__ er,
                             const unsigned* __restrict__ nmax,
                             float* __restrict__ ex, float* __restrict__ nden) {
  const int stride = gridDim.x * blockDim.x;
  for (int t = blockIdx.x * blockDim.x + threadIdx.x; t < MMP * EE; t += stride) {
    const int m = t / EE, e = t - m * EE;
    const size_t idx = (size_t)m * EE + e;
    const int s = src[idx];
    const int d = dst[idx];
#pragma unroll
    for (int hh = 0; hh < HH; ++hh) {
      const float v  = lreluf(el[((size_t)m * NN + s) * HH + hh] +
                              er[((size_t)m * NN + d) * HH + hh]);
      const float mx = fdec(nmax[((size_t)m * NN + d) * HH + hh]);
      const float xv = expf(v - mx);
      ex[idx * HH + hh] = xv;
      atomicAdd(&nden[((size_t)m * NN + d) * HH + hh], xv);
    }
  }
}

// ---- K5: rst[d] += alpha * feat[s]  (wave per edge, coalesced 1KB rows) ---
__global__ void han_edge_aggr(const int* __restrict__ src, const int* __restrict__ dst,
                              const float* __restrict__ ex, const float* __restrict__ nden,
                              const float* __restrict__ feat, float* __restrict__ rst) {
  const int lane  = threadIdx.x & 31;
  const int wpb   = blockDim.x >> 5;
  int wave        = blockIdx.x * wpb + (threadIdx.x >> 5);
  const int nWave = gridDim.x * wpb;
  for (int t = wave; t < MMP * EE; t += nWave) {
    const int m = t / EE, e = t - m * EE;
    const size_t idx = (size_t)m * EE + e;
    const int s = src[idx];
    const int d = dst[idx];
    float a = 0.0f;
    if (lane < HH)
      a = ex[idx * HH + lane] /
          fmaxf(nden[((size_t)m * NN + d) * HH + lane], 1e-9f);
    float alp[HH];
#pragma unroll
    for (int hh = 0; hh < HH; ++hh) alp[hh] = __shfl(a, hh, 32);
    const float* fs = feat + ((size_t)m * NN + s) * HDIM;
    float* rd       = rst  + ((size_t)m * NN + d) * HDIM;
#pragma unroll
    for (int j = 0; j < 8; ++j) {
      const int c = lane + 32 * j;          // head = j>>1
      atomicAdd(&rd[c], alp[j >> 1] * fs[c]);
    }
  }
}

// ---- K6: semantic scores: wacc[m] += sum_n w2 . tanh(elu(rst) w1 + b1) ----
// A fragment built on the fly (fused ELU + bf16 split); B pre-swizzled.
__global__ void han_semantic(const float* __restrict__ rst,
                             const unsigned short* __restrict__ Bhi,
                             const unsigned short* __restrict__ Blo,
                             const float* __restrict__ b1,
                             const float* __restrict__ w2,
                             float* __restrict__ wacc) {
  const int lane  = threadIdx.x & 31;
  const int wpb   = blockDim.x >> 5;
  int wave        = blockIdx.x * wpb + (threadIdx.x >> 5);
  const int nWave = gridDim.x * wpb;
  const int rowT = NN / 16, KT = HDIM / 32, CT = SHID / 16;
  const int total = MMP * rowT * CT;
  const int half = (lane >> 4) & 1;
  for (int t = wave; t < total; t += nWave) {
    const int m   = t / (rowT * CT);
    const int rem = t - m * rowT * CT;
    const int rt  = rem / CT;
    const int ct  = rem - rt * CT;
    const float* Z = rst + ((size_t)m * NN + rt * 16 + (lane & 15)) * HDIM;
    v8f acc = {};
    for (int kt = 0; kt < KT; ++kt) {
      v16us hv, lv;
#pragma unroll
      for (int i = 0; i < 16; ++i) {
        const int j = i >> 1, b = i & 1;
        const int k = kt * 32 + ((j < 4) ? 2 * j : 16 + 2 * (j - 4)) + 8 * half + b;
        unsigned short hb, lb;
        bf16_split(eluf(Z[k]), hb, lb);
        hv[i] = hb; lv[i] = lb;
      }
      const size_t brec = ((size_t)(kt * CT + ct) * 32 + lane) * 16;
      acc = wmma3(acc, __builtin_bit_cast(v16bf, hv), __builtin_bit_cast(v16bf, lv),
                  ldfrag(Bhi + brec), ldfrag(Blo + brec));
    }
    const int col = ct * 16 + (lane & 15);
    const float b1v = b1[col];
    const float w2v = w2[col];
    float wsum = 0.0f;
#pragma unroll
    for (int i = 0; i < 8; ++i) {
      float hid = tanhf(acc[i] + b1v) * w2v;
      hid += __shfl_xor(hid, 1, 16);   // reduce the 16 columns in this half
      hid += __shfl_xor(hid, 2, 16);
      hid += __shfl_xor(hid, 4, 16);
      hid += __shfl_xor(hid, 8, 16);
      wsum += hid;                     // row sum for row i + 8*half
    }
    if ((lane & 15) == 0) atomicAdd(&wacc[m], wsum);
  }
}

// ---- K7: beta = softmax(wacc/N); out = beta0*elu(rst0) + beta1*elu(rst1) --
__global__ void han_final(const float* __restrict__ rst,
                          const float* __restrict__ wacc,
                          float* __restrict__ out) {
  const float w0 = wacc[0] * (1.0f / NN);
  const float w1 = wacc[1] * (1.0f / NN);
  const float mx = fmaxf(w0, w1);
  const float e0 = expf(w0 - mx), e1 = expf(w1 - mx);
  const float inv = 1.0f / (e0 + e1);
  const float b0 = e0 * inv, b1 = e1 * inv;
  const size_t total = (size_t)NN * HDIM;
  const size_t stride = (size_t)gridDim.x * blockDim.x;
  for (size_t i = blockIdx.x * (size_t)blockDim.x + threadIdx.x; i < total; i += stride)
    out[i] = b0 * eluf(rst[i]) + b1 * eluf(rst[total + i]);
}

// ---------------------------------------------------------------------------
extern "C" void kernel_launch(void* const* d_in, const int* in_sizes, int n_in,
                              void* d_out, int out_size, void* d_ws, size_t ws_size,
                              hipStream_t stream) {
  const float* h      = (const float*)d_in[0];   // [N, IN]
  const int*   src    = (const int*)  d_in[1];   // [M, E]
  const int*   dst    = (const int*)  d_in[2];   // [M, E]
  const float* W      = (const float*)d_in[3];   // [M, IN, HD]
  const float* attn_l = (const float*)d_in[4];   // [M, H, D]
  const float* attn_r = (const float*)d_in[5];   // [M, H, D]
  const float* sem_w1 = (const float*)d_in[6];   // [HD, SEM_HID]
  const float* sem_b1 = (const float*)d_in[7];   // [SEM_HID]
  const float* sem_w2 = (const float*)d_in[8];   // [SEM_HID, 1]
  float* out = (float*)d_out;

  char* ws = (char*)d_ws;
  size_t off = 0;
  auto carve = [&](size_t bytes) -> void* {
    void* p = ws + off;
    off += (bytes + 255) & ~(size_t)255;
    return p;
  };
  const size_t hfragN  = (size_t)(NN / 16) * (INF_ / 32) * 32 * 16;   // shorts
  const size_t wfragN  = (size_t)MMP * (INF_ / 32) * (HDIM / 16) * 32 * 16;
  const size_t w1fragN = (size_t)(HDIM / 32) * (SHID / 16) * 32 * 16;

  float*    feat = (float*)   carve((size_t)MMP * NN * HDIM * 4);  // 102.4 MB
  float*    rst  = (float*)   carve((size_t)MMP * NN * HDIM * 4);  // 102.4 MB
  float*    el   = (float*)   carve((size_t)MMP * NN * HH * 4);
  float*    er   = (float*)   carve((size_t)MMP * NN * HH * 4);
  unsigned* nmax = (unsigned*)carve((size_t)MMP * NN * HH * 4);
  float*    nden = (float*)   carve((size_t)MMP * NN * HH * 4);
  float*    ex   = (float*)   carve((size_t)MMP * EE * HH * 4);    // 25.6 MB
  float*    wacc = (float*)   carve(8 * 4);
  unsigned short* hAhi  = (unsigned short*)carve(hfragN * 2);      // 25.6 MB
  unsigned short* hAlo  = (unsigned short*)carve(hfragN * 2);      // 25.6 MB
  unsigned short* wBhi  = (unsigned short*)carve(wfragN * 2);
  unsigned short* wBlo  = (unsigned short*)carve(wfragN * 2);
  unsigned short* w1Bhi = (unsigned short*)carve(w1fragN * 2);
  unsigned short* w1Blo = (unsigned short*)carve(w1fragN * 2);
  (void)ws_size; (void)in_sizes; (void)n_in; (void)out_size;

  // zero accumulators (harness does not re-poison between replays)
  hipMemsetAsync(rst,  0, (size_t)MMP * NN * HDIM * 4, stream);
  hipMemsetAsync(nmax, 0, (size_t)MMP * NN * HH * 4, stream);  // 0 < fenc(finite)
  hipMemsetAsync(nden, 0, (size_t)MMP * NN * HH * 4, stream);
  hipMemsetAsync(wacc, 0, 8 * 4, stream);

  // pre-swizzle operands into WMMA fragment layout (hi/lo bf16 split)
  han_swizzle_a<<<2048, 256, 0, stream>>>(h, hAhi, hAlo, NN, INF_);
  const size_t wfrag1 = wfragN / MMP;
  for (int m = 0; m < MMP; ++m)
    han_swizzle_b<<<64, 256, 0, stream>>>(W + (size_t)m * INF_ * HDIM,
                                          wBhi + m * wfrag1, wBlo + m * wfrag1,
                                          INF_, HDIM);
  han_swizzle_b<<<16, 256, 0, stream>>>(sem_w1, w1Bhi, w1Blo, HDIM, SHID);

  han_feat_gemm  <<<2048, 256, 0, stream>>>(hAhi, hAlo, wBhi, wBlo, feat);
  han_attn_scores<<<2048, 256, 0, stream>>>(feat, attn_l, attn_r, el, er);
  han_edge_max   <<<4096, 256, 0, stream>>>(src, dst, el, er, nmax);
  han_edge_exp   <<<4096, 256, 0, stream>>>(src, dst, el, er, nmax, ex, nden);
  han_edge_aggr  <<<8192, 256, 0, stream>>>(src, dst, ex, nden, feat, rst);
  han_semantic   <<<2048, 256, 0, stream>>>(rst, w1Bhi, w1Blo, sem_b1, sem_w2, wacc);
  han_final      <<<4096, 256, 0, stream>>>(rst, wacc, out);
}
